// MultiHeadAttention_58772332478988
// MI455X (gfx1250) — matrix operand order repaired and checked by reading, not compiled, not run on hardware
//
#include <hip/hip_runtime.h>
#include <hip/hip_bf16.h>
#include <math.h>

// ---------------------------------------------------------------------------
// MI455X / gfx1250 implementation. bf16 WMMA (16x16x32, f32 accum) for all
// GEMM-shaped work; TDM (tensor_load_to_lds) for flash-attention K/V tiles;
// f32 VALU for softmax / local-mix / relation / layernorm.
// ---------------------------------------------------------------------------

typedef __bf16 bf16;
typedef __attribute__((ext_vector_type(16))) __bf16 v16bf;
typedef __attribute__((ext_vector_type(8)))  float  v8f;
typedef __attribute__((ext_vector_type(4)))  int    i32x4;
typedef __attribute__((ext_vector_type(8)))  int    i32x8;

__device__ __forceinline__ v8f zero8() { v8f z = {0.f,0.f,0.f,0.f,0.f,0.f,0.f,0.f}; return z; }

// A-matrix fragment (16x32 bf16): lane = (row = L%16, khalf = L/16),
// element i -> K = kbase + khalf*16 + i
__device__ __forceinline__ v16bf load_frag_a(const bf16* rowPtr, int khalf, int kbase) {
  union { v16bf v; unsigned int u[8]; } r;
  const unsigned int* p0 = (const unsigned int*)(rowPtr + kbase + khalf * 16);
  const unsigned int* p1 = (const unsigned int*)(rowPtr + kbase + 8 + khalf * 16);
#pragma unroll
  for (int j = 0; j < 4; ++j) { r.u[j] = p0[j]; r.u[4 + j] = p1[j]; }
  return r.v;
}

// B-matrix fragment (32x16 bf16): lane = (col = L%16, kgroup = L/16),
// element i -> K = kbase + kgroup*16 + i. colPtr points at tile[col][0]
// of a [N][K]-contiguous LDS tile (K contiguous).
__device__ __forceinline__ v16bf load_frag_b(const bf16* colPtr, int kgroup, int kbase) {
  union { v16bf v; unsigned int u[8]; } r;
  const unsigned int* p = (const unsigned int*)(colPtr + kbase + kgroup * 16);
#pragma unroll
  for (int j = 0; j < 8; ++j) r.u[j] = p[j];
  return r.v;
}

__device__ __forceinline__ v8f wmma_bf16(v16bf a, v16bf b, v8f c) {
  return __builtin_amdgcn_wmma_f32_16x16x32_bf16(false, a, false, b, (short)0, c, false, false);
}

// ---------------------------------------------------------------------------
// TDM: load a 64x64 bf16 tile (row stride 64 elems, contiguous 8KB) into LDS.
// D# per CDNA5 ISA §8.3/8.4: group0 {count=1, lds_addr, global_addr, type=2},
// group1 {data_size=1(2B), tensor_dim0=64, tensor_dim1=2048, tile=64x64,
//         dim0_stride=64}. groups 2/3 zero (2D tensor). This toolchain's
// builtin is the 6-arg variant: (x4, x8, x4, x4, x8, cpol).
// ---------------------------------------------------------------------------
__device__ __forceinline__ void tdm_load_tile_64x64_bf16(unsigned int lds_off,
                                                         unsigned long long gaddr) {
  i32x4 g0;
  g0[0] = 1;                                                   // count = 1
  g0[1] = (int)lds_off;                                        // lds_addr
  g0[2] = (int)(unsigned int)gaddr;                            // global_addr lo
  g0[3] = (int)(((unsigned int)(gaddr >> 32) & 0x01FFFFFFu) | 0x80000000u); // hi | type=2
  i32x8 g1;
  g1[0] = 0x10000;            // workgroup_mask=0, data_size=1 (2 bytes)
  g1[1] = (64 << 16);         // tensor_dim0[15:0]=64 (bits 63:48 here)
  g1[2] = (int)(2048u << 16); // tensor_dim0 hi=0 | tensor_dim1 lo=2048
  g1[3] = (64 << 16);         // tensor_dim1 hi=0 | tile_dim0=64
  g1[4] = 64;                 // tile_dim1=64, tile_dim2=0
  g1[5] = 64;                 // tensor_dim0_stride lo32 = 64
  g1[6] = 0;                  // stride0 hi | stride1 lo
  g1[7] = 0;                  // stride1 hi
  i32x4 z4 = {0, 0, 0, 0};
  i32x8 z8 = {0, 0, 0, 0, 0, 0, 0, 0};
  __builtin_amdgcn_tensor_load_to_lds(g0, g1, z4, z4, z8, 0);
}

// ---------------------------------------------------------------------------
// GEMM: out[r,c] = sum_d A[r,d] * W[c,d]   (A: Mx512 f32, W: 512x512 f32)
// M = 8192, N = K = 512. Block = 128x128 tile, 8 waves, each wave 32x64.
// Optional f32 and/or bf16 outputs.
// ---------------------------------------------------------------------------
__global__ void __launch_bounds__(256)
gemm512(const float* __restrict__ A, const float* __restrict__ W,
        float* __restrict__ outF, bf16* __restrict__ outB) {
  __shared__ __align__(16) bf16 As[128][32];
  __shared__ __align__(16) bf16 Bs[128][32];
  const int tid  = threadIdx.x;
  const int lane = tid & 31, w = tid >> 5;
  const int mBase = blockIdx.y * 128, nBase = blockIdx.x * 128;
  const int wm = w >> 1, wn = w & 1;           // 4 x 2 wave grid
  const int lrow = lane & 15, lhalf = lane >> 4;

  v8f acc[2][4];
#pragma unroll
  for (int sm = 0; sm < 2; ++sm)
#pragma unroll
    for (int sn = 0; sn < 4; ++sn) acc[sm][sn] = zero8();

  for (int kb = 0; kb < 512; kb += 32) {
    // Stage A (128x32) and B (128x32) f32 -> bf16 into LDS.
    for (int i = tid; i < 1024; i += 256) {
      int r = i >> 3, q = i & 7;
      float4 av = *(const float4*)(A + (size_t)(mBase + r) * 512 + kb + q * 4);
      float4 bv = *(const float4*)(W + (size_t)(nBase + r) * 512 + kb + q * 4);
      union { bf16 h[4]; uint2 u2; } pa, pb;
      pa.h[0] = (bf16)av.x; pa.h[1] = (bf16)av.y; pa.h[2] = (bf16)av.z; pa.h[3] = (bf16)av.w;
      pb.h[0] = (bf16)bv.x; pb.h[1] = (bf16)bv.y; pb.h[2] = (bf16)bv.z; pb.h[3] = (bf16)bv.w;
      *(uint2*)&As[r][q * 4] = pa.u2;
      *(uint2*)&Bs[r][q * 4] = pb.u2;
    }
    if (kb + 32 < 512) {   // CDNA5 prefetch of next K slab
      __builtin_prefetch(A + (size_t)(mBase + wm * 32) * 512 + kb + 32, 0, 1);
      __builtin_prefetch(W + (size_t)(nBase + wn * 64) * 512 + kb + 32, 0, 1);
    }
    __syncthreads();

    v16bf af[2], bfr[4];
#pragma unroll
    for (int sm = 0; sm < 2; ++sm)
      af[sm] = load_frag_a(&As[wm * 32 + sm * 16 + lrow][0], lhalf, 0);
#pragma unroll
    for (int sn = 0; sn < 4; ++sn)
      bfr[sn] = load_frag_b(&Bs[wn * 64 + sn * 16 + lrow][0], lhalf, 0);
#pragma unroll
    for (int sm = 0; sm < 2; ++sm)
#pragma unroll
      for (int sn = 0; sn < 4; ++sn)
        acc[sm][sn] = wmma_bf16(af[sm], bfr[sn], acc[sm][sn]);
    __syncthreads();
  }

  // Epilogue: C layout -> row = vgpr + (L/16)*8, col = L%16.
#pragma unroll
  for (int sm = 0; sm < 2; ++sm)
#pragma unroll
    for (int sn = 0; sn < 4; ++sn)
#pragma unroll
      for (int v = 0; v < 8; ++v) {
        int row = mBase + wm * 32 + sm * 16 + lhalf * 8 + v;
        int col = nBase + wn * 64 + sn * 16 + lrow;
        float val = acc[sm][sn][v];
        if (outF) outF[(size_t)row * 512 + col] = val;
        if (outB) outB[(size_t)row * 512 + col] = (bf16)val;
      }
}

// ---------------------------------------------------------------------------
// local_mix on the reinterpreted (m,H,k,64) view. 8-row causal window with
// ones padding; out[i] = sum_{o=0..7} (dot(x[i], prev[o])/8) * prev[o].
// Block: one (m,h) x 64-row chunk; 8 waves x 8 rows. Emits bf16.
// ---------------------------------------------------------------------------
__global__ void __launch_bounds__(256)
localmix(const float* __restrict__ proj, bf16* __restrict__ outB) {
  __shared__ float Ls[71 * 64];
  const int tid = threadIdx.x, lane = tid & 31, w = tid >> 5;
  const int mh = blockIdx.y;
  const int rowChunk = blockIdx.x * 64;
  const size_t base = (size_t)mh * 2048 * 64;

  for (int i = tid; i < 71 * 64; i += 256) {
    int r = i >> 6, d = i & 63;
    int g = rowChunk - 7 + r;
    Ls[i] = (g < 0) ? 1.0f : proj[base + (size_t)g * 64 + d];
  }
  __syncthreads();

  const int d0 = lane * 2;
  for (int rr = 0; rr < 8; ++rr) {
    int lr = w * 8 + rr;
    const float* x = &Ls[(lr + 7) * 64];
    float x0 = x[d0], x1 = x[d0 + 1];
    float o0 = 0.f, o1 = 0.f;
#pragma unroll
    for (int o = 0; o < 8; ++o) {
      const float* p = &Ls[(lr + o) * 64];
      float p0 = p[d0], p1 = p[d0 + 1];
      float part = x0 * p0 + x1 * p1;
      part += __shfl_xor(part, 16);
      part += __shfl_xor(part, 8);
      part += __shfl_xor(part, 4);
      part += __shfl_xor(part, 2);
      part += __shfl_xor(part, 1);
      float t = part * 0.125f;                 // 1/sqrt(64)
      o0 += t * p0; o1 += t * p1;
    }
    union { bf16 h[2]; unsigned int u; } pk;
    pk.h[0] = (bf16)o0; pk.h[1] = (bf16)o1;
    *(unsigned int*)&outB[base + (size_t)(rowChunk + lr) * 64 + d0] = pk.u;
  }
}

// ---------------------------------------------------------------------------
// Flash SDPA: per (m,h) batch, 128 q-rows per block, 64-row K/V tiles
// streamed via TDM into LDS. Temperature = 1/64 on scores.
// ---------------------------------------------------------------------------
__global__ void __launch_bounds__(256)
flash_sdpa(const bf16* __restrict__ Qb, const bf16* __restrict__ Kb,
           const bf16* __restrict__ Vb, float* __restrict__ O) {
  __shared__ __align__(16) bf16 Qs[128][64];
  __shared__ __align__(16) bf16 Ks[64][64];
  __shared__ __align__(16) bf16 Vs[64][64];
  __shared__ __align__(16) bf16 Ps[8][16][64];

  const int tid = threadIdx.x, lane = tid & 31, w = tid >> 5;
  const int lrow = lane & 15, lhalf = lane >> 4;
  const int mh = blockIdx.y;
  const int qBase = blockIdx.x * 128;
  const size_t sbase = (size_t)mh * 2048 * 64;

  { // Q tile (already bf16) -> LDS
    const uint4* gq = (const uint4*)(Qb + sbase + (size_t)qBase * 64);
    uint4* sq = (uint4*)&Qs[0][0];
    for (int i = tid; i < 128 * 64 * 2 / 16; i += 256) sq[i] = gq[i];
  }
  __syncthreads();

  v8f accO[4];
  float mrow[8], lrowv[8];
#pragma unroll
  for (int t = 0; t < 4; ++t) accO[t] = zero8();
#pragma unroll
  for (int v = 0; v < 8; ++v) { mrow[v] = -INFINITY; lrowv[v] = 0.f; }

  const unsigned long long kAddr = (unsigned long long)(const void*)(Kb + sbase);
  const unsigned long long vAddr = (unsigned long long)(const void*)(Vb + sbase);
  const unsigned int ldsK = (unsigned int)(size_t)&Ks[0][0];
  const unsigned int ldsV = (unsigned int)(size_t)&Vs[0][0];

  for (int kb = 0; kb < 2048; kb += 64) {
    __syncthreads();                               // prior tile fully consumed
    if (tid < 32) {                                // wave 0 drives the TDM
      tdm_load_tile_64x64_bf16(ldsK, kAddr + (unsigned long long)kb * 64 * 2);
      tdm_load_tile_64x64_bf16(ldsV, vAddr + (unsigned long long)kb * 64 * 2);
      __builtin_amdgcn_s_wait_tensorcnt(0);
    }
    __syncthreads();

    // S = Q * K^T  (16 q-rows per wave x 64 key cols), f32 accum
    v8f accS[4];
#pragma unroll
    for (int t = 0; t < 4; ++t) accS[t] = zero8();
    v16bf qf0 = load_frag_a(&Qs[w * 16 + lrow][0], lhalf, 0);
    v16bf qf1 = load_frag_a(&Qs[w * 16 + lrow][0], lhalf, 32);
#pragma unroll
    for (int t = 0; t < 4; ++t) {
      const bf16* kcol = &Ks[t * 16 + lrow][0];
      accS[t] = wmma_bf16(qf0, load_frag_b(kcol, lhalf, 0),  accS[t]);
      accS[t] = wmma_bf16(qf1, load_frag_b(kcol, lhalf, 32), accS[t]);
    }

    // Online softmax (temperature 1/64). Row stats reduce over the 16-lane
    // half that owns the row (shfl masks 1,2,4,8).
    float pbuf[4][8];
#pragma unroll
    for (int v = 0; v < 8; ++v) {
      float mv = -INFINITY;
#pragma unroll
      for (int t = 0; t < 4; ++t) {
        accS[t][v] *= 0.015625f;                   // 1/64
        mv = fmaxf(mv, accS[t][v]);
      }
      mv = fmaxf(mv, __shfl_xor(mv, 1));
      mv = fmaxf(mv, __shfl_xor(mv, 2));
      mv = fmaxf(mv, __shfl_xor(mv, 4));
      mv = fmaxf(mv, __shfl_xor(mv, 8));
      float mnew = fmaxf(mrow[v], mv);
      float sc = __expf(mrow[v] - mnew);
      mrow[v] = mnew;
      lrowv[v] *= sc;
      accO[0][v] *= sc; accO[1][v] *= sc; accO[2][v] *= sc; accO[3][v] *= sc;
      float ps = 0.f;
#pragma unroll
      for (int t = 0; t < 4; ++t) {
        float p = __expf(accS[t][v] - mnew);
        pbuf[t][v] = p; ps += p;
      }
      ps += __shfl_xor(ps, 1);
      ps += __shfl_xor(ps, 2);
      ps += __shfl_xor(ps, 4);
      ps += __shfl_xor(ps, 8);
      lrowv[v] += ps;
    }

    // P (C layout) -> LDS -> A layout for P*V.
#pragma unroll
    for (int t = 0; t < 4; ++t)
#pragma unroll
      for (int v = 0; v < 8; ++v)
        Ps[w][lhalf * 8 + v][t * 16 + lrow] = (bf16)pbuf[t][v];
    asm volatile("s_wait_dscnt 0" ::: "memory");

    v16bf pf0 = load_frag_a(&Ps[w][lrow][0], lhalf, 0);
    v16bf pf1 = load_frag_a(&Ps[w][lrow][0], lhalf, 32);
#pragma unroll
    for (int t = 0; t < 4; ++t) {
      union { v16bf v; bf16 h[16]; } bv0, bv1;
      int coln = t * 16 + lrow;
      int kg = lhalf * 16;
#pragma unroll
      for (int i = 0; i < 16; ++i) {
        bv0.h[i] = Vs[kg + i][coln];               // K = kgroup*16 + i
        bv1.h[i] = Vs[32 + kg + i][coln];          // K = 32 + ...
      }
      accO[t] = wmma_bf16(pf0, bv0.v, accO[t]);
      accO[t] = wmma_bf16(pf1, bv1.v, accO[t]);
    }
  }

  // Normalize and store (f32).
#pragma unroll
  for (int t = 0; t < 4; ++t)
#pragma unroll
    for (int v = 0; v < 8; ++v) {
      int row = qBase + w * 16 + lhalf * 8 + v;
      int col = t * 16 + lrow;
      O[sbase + (size_t)row * 64 + col] = accO[t][v] / lrowv[v];
    }
}

// ---------------------------------------------------------------------------
// Relation SDPA: 16384 independent 4x4 attentions on contiguous 256-float
// groups (group g at flat offset g*256). In place. One wave per group.
// ---------------------------------------------------------------------------
__global__ void __launch_bounds__(256)
relation_sdpa(float* __restrict__ O) {
  const int lane = threadIdx.x & 31, w = threadIdx.x >> 5;
  const int g = blockIdx.x * 8 + w;
  float* base = O + (size_t)g * 256;
  const int d0 = lane * 2;

  float x[4][2];
#pragma unroll
  for (int i = 0; i < 4; ++i) { x[i][0] = base[i * 64 + d0]; x[i][1] = base[i * 64 + d0 + 1]; }

  float s[4][4];
#pragma unroll
  for (int i = 0; i < 4; ++i)
#pragma unroll
    for (int j = 0; j < 4; ++j) {
      float part = x[i][0] * x[j][0] + x[i][1] * x[j][1];
      part += __shfl_xor(part, 16);
      part += __shfl_xor(part, 8);
      part += __shfl_xor(part, 4);
      part += __shfl_xor(part, 2);
      part += __shfl_xor(part, 1);
      s[i][j] = part * 0.015625f;                  // q / D_F
    }

  float a[4][4];
#pragma unroll
  for (int i = 0; i < 4; ++i) {
    float mx = fmaxf(fmaxf(s[i][0], s[i][1]), fmaxf(s[i][2], s[i][3]));
    float den = 0.f;
#pragma unroll
    for (int j = 0; j < 4; ++j) { a[i][j] = __expf(s[i][j] - mx); den += a[i][j]; }
    float inv = 1.f / den;
#pragma unroll
    for (int j = 0; j < 4; ++j) a[i][j] *= inv;
  }

#pragma unroll
  for (int i = 0; i < 4; ++i) {
    float o0 = 0.f, o1 = 0.f;
#pragma unroll
    for (int j = 0; j < 4; ++j) { o0 += a[i][j] * x[j][0]; o1 += a[i][j] * x[j][1]; }
    base[i * 64 + d0] = o0; base[i * 64 + d0 + 1] = o1;
  }
}

// ---------------------------------------------------------------------------
// residual + LayerNorm over 512-wide rows: out = LN(query + Y2)*scale + bias
// ---------------------------------------------------------------------------
__global__ void __launch_bounds__(256)
residual_ln(const float* __restrict__ Y2, const float* __restrict__ Q,
            const float* __restrict__ scale, const float* __restrict__ bias,
            float* __restrict__ out) {
  __shared__ float rs[8], rs2[8];
  const int row = blockIdx.x, tid = threadIdx.x, lane = tid & 31, w = tid >> 5;
  const size_t b = (size_t)row * 512 + tid * 2;
  float a0 = Q[b] + Y2[b], a1 = Q[b + 1] + Y2[b + 1];
  float s = a0 + a1, s2 = a0 * a0 + a1 * a1;
#pragma unroll
  for (int m = 16; m >= 1; m >>= 1) { s += __shfl_xor(s, m); s2 += __shfl_xor(s2, m); }
  if (lane == 0) { rs[w] = s; rs2[w] = s2; }
  __syncthreads();
  float ts = 0.f, ts2 = 0.f;
#pragma unroll
  for (int i = 0; i < 8; ++i) { ts += rs[i]; ts2 += rs2[i]; }
  float mu = ts * (1.f / 512.f);
  float var = ts2 * (1.f / 512.f) - mu * mu;
  float inv = rsqrtf(var + 1e-5f);
  int c = tid * 2;
  out[b]     = (a0 - mu) * inv * scale[c]     + bias[c];
  out[b + 1] = (a1 - mu) * inv * scale[c + 1] + bias[c + 1];
}

// ---------------------------------------------------------------------------
// Host: dispatch pipeline. Workspace layout (104 MB):
//   projq f32 16MB | projk f32 16MB | Qb bf16 8MB | Kb bf16 8MB | Vb bf16 8MB
//   O f32 16MB | Y1 f32 16MB | Y2 f32 16MB
// ---------------------------------------------------------------------------
extern "C" void kernel_launch(void* const* d_in, const int* in_sizes, int n_in,
                              void* d_out, int out_size, void* d_ws, size_t ws_size,
                              hipStream_t stream) {
  const float* q        = (const float*)d_in[0];
  const float* kkey     = (const float*)d_in[1];
  const float* vval     = (const float*)d_in[2];
  const float* Wq       = (const float*)d_in[3];
  const float* Wk       = (const float*)d_in[4];
  const float* Wv       = (const float*)d_in[5];
  const float* Wfc      = (const float*)d_in[6];
  const float* ln_scale = (const float*)d_in[7];
  const float* ln_bias  = (const float*)d_in[8];

  constexpr size_t SZ_F32  = 8192ull * 512 * 4;   // 16 MB
  constexpr size_t SZ_BF16 = 8192ull * 512 * 2;   //  8 MB
  char* ws = (char*)d_ws;
  float* projq = (float*)(ws);
  float* projk = (float*)(ws + SZ_F32);
  bf16*  Qb    = (bf16*)(ws + 2 * SZ_F32);
  bf16*  Kb    = (bf16*)(ws + 2 * SZ_F32 + SZ_BF16);
  bf16*  Vb    = (bf16*)(ws + 2 * SZ_F32 + 2 * SZ_BF16);
  float* O     = (float*)(ws + 2 * SZ_F32 + 3 * SZ_BF16);
  float* Y1    = (float*)(ws + 3 * SZ_F32 + 3 * SZ_BF16);
  float* Y2    = (float*)(ws + 4 * SZ_F32 + 3 * SZ_BF16);

  dim3 blk(256);
  dim3 gemmGrid(4, 64);      // 512/128 N-blocks x 8192/128 M-blocks

  gemm512<<<gemmGrid, blk, 0, stream>>>(q,    Wq, projq, nullptr);
  gemm512<<<gemmGrid, blk, 0, stream>>>(kkey, Wk, projk, nullptr);
  gemm512<<<gemmGrid, blk, 0, stream>>>(vval, Wv, nullptr, Vb);

  localmix<<<dim3(32, 32), blk, 0, stream>>>(projq, Qb);
  localmix<<<dim3(32, 32), blk, 0, stream>>>(projk, Kb);

  flash_sdpa<<<dim3(16, 32), blk, 0, stream>>>(Qb, Kb, Vb, O);

  relation_sdpa<<<dim3(2048), blk, 0, stream>>>(O);

  gemm512<<<gemmGrid, blk, 0, stream>>>(O,  Wfc, Y1, nullptr);
  gemm512<<<gemmGrid, blk, 0, stream>>>(Y1, Wfc, Y2, nullptr);

  residual_ln<<<dim3(8192), blk, 0, stream>>>(Y2, q, ln_scale, ln_bias, (float*)d_out);
}
